// MultiHeadAttention_80599356276988
// MI455X (gfx1250) — matrix operand order repaired and checked by reading, not compiled, not run on hardware
//
#include <hip/hip_runtime.h>
#include <hip/hip_bf16.h>
#include <math.h>

#define B_ 2
#define T_ 2048
#define C_ 1024
#define H_ 16
#define D_ 64

typedef __attribute__((ext_vector_type(16))) __bf16 v16bf;
typedef __attribute__((ext_vector_type(8)))  float  v8f;
typedef int vi4 __attribute__((vector_size(16)));   // int4 vector (16B)

union Frag {
  v16bf v;
  uint4 q[2];
};

#if defined(__HIP_DEVICE_COMPILE__) && defined(__has_builtin)
#if __has_builtin(__builtin_amdgcn_global_load_async_to_lds_b128) && \
    __has_builtin(__builtin_amdgcn_s_wait_asynccnt)
#define HAVE_ASYNC_LDS 1
#endif
#endif
#ifndef HAVE_ASYNC_LDS
#define HAVE_ASYNC_LDS 0
#endif

__device__ inline void stage16(const __bf16* g, unsigned short* l) {
#if HAVE_ASYNC_LDS
  __builtin_amdgcn_global_load_async_to_lds_b128(
      (__attribute__((address_space(1))) vi4*)g,
      (__attribute__((address_space(3))) vi4*)l, 0, 0);
#else
  *(uint4*)l = *(const uint4*)g;
#endif
}

#if HAVE_ASYNC_LDS
#define WAIT_ASYNC(n) __builtin_amdgcn_s_wait_asynccnt(n)
#else
#define WAIT_ASYNC(n)
#endif

__device__ inline unsigned short f2bf(float x) {
  union { __bf16 b; unsigned short u; } cv;
  cv.b = (__bf16)x;
  return cv.u;
}

__device__ inline float redmax16(float x) {
#pragma unroll
  for (int m = 1; m < 16; m <<= 1) x = fmaxf(x, __shfl_xor(x, m, 32));
  return x;
}
__device__ inline float redsum16(float x) {
#pragma unroll
  for (int m = 1; m < 16; m <<= 1) x += __shfl_xor(x, m, 32);
  return x;
}

// ---------------------------------------------------------------- convert
__global__ void mha_cvt_bf16(const float* __restrict__ in,
                             __bf16* __restrict__ out, int n) {
  int i = blockIdx.x * blockDim.x + threadIdx.x;
  int stride = gridDim.x * blockDim.x;
  for (; i < n; i += stride) out[i] = (__bf16)in[i];
}

// ---------------------------------------------------------------- QKV GEMM
// Block: 32 rows of X shared by 8 waves; A-tile staged to LDS via async DMA,
// double-buffered. Each wave computes a 32x32 output tile (2x2 WMMA tiles).
// out rows n in [0,3072): p = n/1024 (0:Q 1:K 2:V), h = (n%1024)/64, d = n%64
__global__ __launch_bounds__(256) void mha_qkv_gemm(
    const __bf16* __restrict__ Xb,   // [B*T, C]
    const __bf16* __restrict__ Wb,   // [3*C, C]
    __bf16* __restrict__ Qb,         // [B,H,T,D]
    __bf16* __restrict__ Kb,         // [B,H,T,D]
    __bf16* __restrict__ VTb) {      // [B,H,D,T]
  __shared__ unsigned short sA[2][32 * 32];   // 2 x 2KB A-tile buffers
  const int tid  = threadIdx.x;
  const int lane = tid & 31;
  const int l15  = lane & 15;
  const int hh   = lane >> 4;
  const int wave = tid >> 5;
  const int r0 = blockIdx.x * 32;               // row tile in [0, B*T)
  const int n0 = (blockIdx.y * 8 + wave) * 32;  // col tile in [0, 3072)

  // loader role: threads 0..127 move the 32x32 bf16 A tile (2KB = 128x16B)
  const int lrow = tid >> 2;
  const int lk   = (tid & 3) * 8;
  const __bf16* a_base = Xb + (size_t)r0 * C_;

  const __bf16* brow0 = Wb + (size_t)(n0 + l15) * C_;
  const __bf16* brow1 = Wb + (size_t)(n0 + 16 + l15) * C_;

  auto stage = [&](int k, int buf) {
    if (tid < 128)
      stage16(a_base + (size_t)lrow * C_ + k + lk, &sA[buf][lrow * 32 + lk]);
  };

  v8f acc[2][2] = {};
  stage(0, 0);
  int k = 0;
  for (; k < C_ - 32; k += 32) {
    const int buf = (k >> 5) & 1;
    stage(k + 32, buf ^ 1);          // prefetch next tile (stays in flight)
    WAIT_ASYNC(1);                    // current tile resident
    __syncthreads();

    Frag a0, a1, b0, b1;
    const unsigned short* sb = sA[buf];
    a0.q[0] = *(const uint4*)(sb + l15 * 32 + hh * 8);
    a0.q[1] = *(const uint4*)(sb + l15 * 32 + 16 + hh * 8);
    a1.q[0] = *(const uint4*)(sb + (16 + l15) * 32 + hh * 8);
    a1.q[1] = *(const uint4*)(sb + (16 + l15) * 32 + 16 + hh * 8);
    __builtin_prefetch(brow0 + k + 128, 0, 1);
    __builtin_prefetch(brow1 + k + 128, 0, 1);
    b0.q[0] = *(const uint4*)(brow0 + k + hh * 8);
    b0.q[1] = *(const uint4*)(brow0 + k + 16 + hh * 8);
    b1.q[0] = *(const uint4*)(brow1 + k + hh * 8);
    b1.q[1] = *(const uint4*)(brow1 + k + 16 + hh * 8);

    acc[0][0] = __builtin_amdgcn_wmma_f32_16x16x32_bf16(false, a0.v, false, b0.v, (short)0, acc[0][0], false, false);
    acc[0][1] = __builtin_amdgcn_wmma_f32_16x16x32_bf16(false, a0.v, false, b1.v, (short)0, acc[0][1], false, false);
    acc[1][0] = __builtin_amdgcn_wmma_f32_16x16x32_bf16(false, a1.v, false, b0.v, (short)0, acc[1][0], false, false);
    acc[1][1] = __builtin_amdgcn_wmma_f32_16x16x32_bf16(false, a1.v, false, b1.v, (short)0, acc[1][1], false, false);
    __syncthreads();                  // all reads done before re-staging buf
  }
  {                                   // final (peeled) iteration
    const int buf = (k >> 5) & 1;
    WAIT_ASYNC(0);
    __syncthreads();
    Frag a0, a1, b0, b1;
    const unsigned short* sb = sA[buf];
    a0.q[0] = *(const uint4*)(sb + l15 * 32 + hh * 8);
    a0.q[1] = *(const uint4*)(sb + l15 * 32 + 16 + hh * 8);
    a1.q[0] = *(const uint4*)(sb + (16 + l15) * 32 + hh * 8);
    a1.q[1] = *(const uint4*)(sb + (16 + l15) * 32 + 16 + hh * 8);
    b0.q[0] = *(const uint4*)(brow0 + k + hh * 8);
    b0.q[1] = *(const uint4*)(brow0 + k + 16 + hh * 8);
    b1.q[0] = *(const uint4*)(brow1 + k + hh * 8);
    b1.q[1] = *(const uint4*)(brow1 + k + 16 + hh * 8);
    acc[0][0] = __builtin_amdgcn_wmma_f32_16x16x32_bf16(false, a0.v, false, b0.v, (short)0, acc[0][0], false, false);
    acc[0][1] = __builtin_amdgcn_wmma_f32_16x16x32_bf16(false, a0.v, false, b1.v, (short)0, acc[0][1], false, false);
    acc[1][0] = __builtin_amdgcn_wmma_f32_16x16x32_bf16(false, a1.v, false, b0.v, (short)0, acc[1][0], false, false);
    acc[1][1] = __builtin_amdgcn_wmma_f32_16x16x32_bf16(false, a1.v, false, b1.v, (short)0, acc[1][1], false, false);
  }

  const int b_ = r0 / T_;
  const int t0 = r0 % T_;
#pragma unroll
  for (int ni = 0; ni < 2; ni++) {
    const int nsub = n0 + ni * 16;
    const int p   = nsub >> 10;
    const int rem = nsub & 1023;
    const int h   = rem >> 6;
    const int d0  = rem & 63;
    const size_t bh = (size_t)b_ * H_ + h;
#pragma unroll
    for (int mi = 0; mi < 2; mi++) {
      const int tb = t0 + mi * 16;
      if (p < 2) {                    // Q or K: [B,H,T,D]
        __bf16* dst = (p == 0) ? Qb : Kb;
#pragma unroll
        for (int i = 0; i < 8; i++)
          dst[(bh * T_ + tb + i + 8 * hh) * D_ + d0 + l15] = (__bf16)acc[mi][ni][i];
      } else {                        // V transposed: [B,H,D,T]
        unsigned short pk[8];
#pragma unroll
        for (int i = 0; i < 8; i++) pk[i] = f2bf(acc[mi][ni][i]);
        uint4 o;
        o.x = (unsigned)pk[0] | ((unsigned)pk[1] << 16);
        o.y = (unsigned)pk[2] | ((unsigned)pk[3] << 16);
        o.z = (unsigned)pk[4] | ((unsigned)pk[5] << 16);
        o.w = (unsigned)pk[6] | ((unsigned)pk[7] << 16);
        *(uint4*)(VTb + (bh * D_ + d0 + l15) * T_ + tb + 8 * hh) = o;
      }
    }
  }
}

// ---------------------------------------------------------------- flash attention
// One wave = 32 query rows (2 m-tiles) so each K/V fragment feeds 2 WMMAs.
__global__ __launch_bounds__(256) void mha_flash(
    const __bf16* __restrict__ Qb,   // [B,H,T,D]
    const __bf16* __restrict__ Kb,   // [B,H,T,D]
    const __bf16* __restrict__ VTb,  // [B,H,D,T]
    __bf16* __restrict__ Ob) {       // [B*T, C] (heads concatenated)
  __shared__ uint4 ldsP[8 * 128];    // per-wave 2KB P staging
  const int lane = threadIdx.x & 31;
  const int l15  = lane & 15;
  const int hh   = lane >> 4;
  const int wave = threadIdx.x >> 5;
  unsigned short* lp = (unsigned short*)(ldsP + wave * 128);

  const int job   = blockIdx.x * 8 + wave;   // [0, B*H*T/32)
  const int mtile = job & 63;
  const int bhid  = job >> 6;                // [0, B*H)
  const int h     = bhid & (H_ - 1);
  const int b_    = bhid >> 4;
  const int t0    = mtile * 32;
  const size_t bhT = (size_t)bhid * T_;
  const float SCALE = 0.03125f;              // C^-0.5 = 1/32

  Frag qa[2][2];
#pragma unroll
  for (int mt = 0; mt < 2; mt++) {
    const __bf16* qrow = Qb + (bhT + t0 + mt * 16 + l15) * D_;
    qa[mt][0].q[0] = *(const uint4*)(qrow + hh * 8);
    qa[mt][0].q[1] = *(const uint4*)(qrow + 16 + hh * 8);
    qa[mt][1].q[0] = *(const uint4*)(qrow + 32 + hh * 8);
    qa[mt][1].q[1] = *(const uint4*)(qrow + 48 + hh * 8);
  }

  v8f o[2][4] = {};
  float mi[2][8], li[2][8];
#pragma unroll
  for (int mt = 0; mt < 2; mt++)
#pragma unroll
    for (int i = 0; i < 8; i++) { mi[mt][i] = -__builtin_inff(); li[mt][i] = 0.f; }

  const int jmax = (t0 + 31) >> 5;
  for (int j = 0; j <= jmax; j++) {
    // ---- K fragments for 32 keys (shared by both m-tiles)
    Frag bk[2][2];
#pragma unroll
    for (int tt = 0; tt < 2; tt++) {
      const __bf16* kr = Kb + (bhT + j * 32 + tt * 16 + l15) * D_;
      bk[tt][0].q[0] = *(const uint4*)(kr + hh * 8);
      bk[tt][0].q[1] = *(const uint4*)(kr + 16 + hh * 8);
      bk[tt][1].q[0] = *(const uint4*)(kr + 32 + hh * 8);
      bk[tt][1].q[1] = *(const uint4*)(kr + 48 + hh * 8);
    }
    // ---- S = Q * K^T : 8 WMMAs for a 32x32 score block
    v8f s[2][2] = {};
#pragma unroll
    for (int mt = 0; mt < 2; mt++)
#pragma unroll
      for (int tt = 0; tt < 2; tt++) {
        s[mt][tt] = __builtin_amdgcn_wmma_f32_16x16x32_bf16(false, qa[mt][0].v, false, bk[tt][0].v, (short)0, s[mt][tt], false, false);
        s[mt][tt] = __builtin_amdgcn_wmma_f32_16x16x32_bf16(false, qa[mt][1].v, false, bk[tt][1].v, (short)0, s[mt][tt], false, false);
      }

    // ---- causal mask + online softmax + P -> LDS (transpose to A-layout)
    const int scol0 = j * 32 + l15;
    const int scol1 = scol0 + 16;
#pragma unroll
    for (int mt = 0; mt < 2; mt++) {
#pragma unroll
      for (int i = 0; i < 8; i++) {
        const int trow = t0 + mt * 16 + i + 8 * hh;
        float v0 = s[mt][0][i] * SCALE;
        float v1 = s[mt][1][i] * SCALE;
        if (scol0 > trow) v0 = -__builtin_inff();
        if (scol1 > trow) v1 = -__builtin_inff();
        float cm = redmax16(fmaxf(v0, v1));
        float mn = fmaxf(mi[mt][i], cm);
        float alpha = __expf(mi[mt][i] - mn);
        float e0 = __expf(v0 - mn);
        float e1 = __expf(v1 - mn);
        float rs = redsum16(e0 + e1);
        li[mt][i] = li[mt][i] * alpha + rs;
        mi[mt][i] = mn;
        o[mt][0][i] *= alpha; o[mt][1][i] *= alpha;
        o[mt][2][i] *= alpha; o[mt][3][i] *= alpha;
        const int m = i + 8 * hh;
        lp[mt * 512 + m * 32 + l15]      = f2bf(e0);
        lp[mt * 512 + m * 32 + 16 + l15] = f2bf(e1);
      }
    }
    Frag pa[2];
#pragma unroll
    for (int mt = 0; mt < 2; mt++) {
      pa[mt].q[0] = *(const uint4*)(lp + mt * 512 + l15 * 32 + hh * 8);
      pa[mt].q[1] = *(const uint4*)(lp + mt * 512 + l15 * 32 + 16 + hh * 8);
    }

    // ---- O += P * V : 8 WMMAs, V fragments shared by both m-tiles
    const __bf16* vrow = VTb + ((size_t)bhid * D_ + l15) * T_ + j * 32 + hh * 8;
#pragma unroll
    for (int nn = 0; nn < 4; nn++) {
      Frag bv;
      bv.q[0] = *(const uint4*)(vrow + (size_t)nn * 16 * T_);
      bv.q[1] = *(const uint4*)(vrow + (size_t)nn * 16 * T_ + 16);
      o[0][nn] = __builtin_amdgcn_wmma_f32_16x16x32_bf16(false, pa[0].v, false, bv.v, (short)0, o[0][nn], false, false);
      o[1][nn] = __builtin_amdgcn_wmma_f32_16x16x32_bf16(false, pa[1].v, false, bv.v, (short)0, o[1][nn], false, false);
    }
  }

  // ---- normalize, write concatenated-head output (bf16)
#pragma unroll
  for (int mt = 0; mt < 2; mt++)
#pragma unroll
    for (int i = 0; i < 8; i++) {
      const float inv = 1.0f / li[mt][i];
      __bf16* orow = Ob + (size_t)(b_ * T_ + t0 + mt * 16 + i + 8 * hh) * C_ + h * D_;
      orow[0 * 16 + l15] = (__bf16)(o[mt][0][i] * inv);
      orow[1 * 16 + l15] = (__bf16)(o[mt][1][i] * inv);
      orow[2 * 16 + l15] = (__bf16)(o[mt][2][i] * inv);
      orow[3 * 16 + l15] = (__bf16)(o[mt][3][i] * inv);
    }
}

// ---------------------------------------------------------------- output projection
__global__ __launch_bounds__(256) void mha_out_proj(
    const __bf16* __restrict__ Ob,   // [B*T, C]
    const __bf16* __restrict__ Wpb,  // [C, C] (row = out, col = in)
    const float* __restrict__ bp,    // [C]
    float* __restrict__ Y) {         // [B*T, C]
  const int lane = threadIdx.x & 31;
  const int l15  = lane & 15;
  const int hh   = lane >> 4;
  const int wave = threadIdx.x >> 5;
  const int r0 = blockIdx.x * 32;
  const int n0 = (blockIdx.y * 8 + wave) * 32;

  const __bf16* ar0 = Ob  + (size_t)(r0 + l15) * C_;
  const __bf16* ar1 = Ob  + (size_t)(r0 + 16 + l15) * C_;
  const __bf16* br0 = Wpb + (size_t)(n0 + l15) * C_;
  const __bf16* br1 = Wpb + (size_t)(n0 + 16 + l15) * C_;

  v8f acc[2][2] = {};
  for (int k = 0; k < C_; k += 32) {
    __builtin_prefetch(ar0 + k + 128, 0, 1);
    __builtin_prefetch(br0 + k + 128, 0, 1);
    Frag a0, a1, b0, b1;
    a0.q[0] = *(const uint4*)(ar0 + k + hh * 8);
    a0.q[1] = *(const uint4*)(ar0 + k + 16 + hh * 8);
    a1.q[0] = *(const uint4*)(ar1 + k + hh * 8);
    a1.q[1] = *(const uint4*)(ar1 + k + 16 + hh * 8);
    b0.q[0] = *(const uint4*)(br0 + k + hh * 8);
    b0.q[1] = *(const uint4*)(br0 + k + 16 + hh * 8);
    b1.q[0] = *(const uint4*)(br1 + k + hh * 8);
    b1.q[1] = *(const uint4*)(br1 + k + 16 + hh * 8);
    acc[0][0] = __builtin_amdgcn_wmma_f32_16x16x32_bf16(false, a0.v, false, b0.v, (short)0, acc[0][0], false, false);
    acc[0][1] = __builtin_amdgcn_wmma_f32_16x16x32_bf16(false, a0.v, false, b1.v, (short)0, acc[0][1], false, false);
    acc[1][0] = __builtin_amdgcn_wmma_f32_16x16x32_bf16(false, a1.v, false, b0.v, (short)0, acc[1][0], false, false);
    acc[1][1] = __builtin_amdgcn_wmma_f32_16x16x32_bf16(false, a1.v, false, b1.v, (short)0, acc[1][1], false, false);
  }

  const float bias0 = bp[n0 + l15];
  const float bias1 = bp[n0 + 16 + l15];
#pragma unroll
  for (int mi = 0; mi < 2; mi++)
#pragma unroll
    for (int i = 0; i < 8; i++) {
      float* yrow = Y + (size_t)(r0 + mi * 16 + i + 8 * hh) * C_;
      yrow[n0 + l15]      = acc[mi][0][i] + bias0;
      yrow[n0 + 16 + l15] = acc[mi][1][i] + bias1;
    }
}

// ---------------------------------------------------------------- launch
extern "C" void kernel_launch(void* const* d_in, const int* in_sizes, int n_in,
                              void* d_out, int out_size, void* d_ws, size_t ws_size,
                              hipStream_t stream) {
  (void)in_sizes; (void)n_in; (void)out_size; (void)ws_size;
  const float* X  = (const float*)d_in[0];
  const float* Wq = (const float*)d_in[1];
  const float* Wk = (const float*)d_in[2];
  const float* Wv = (const float*)d_in[3];
  const float* Wp = (const float*)d_in[4];
  const float* bp = (const float*)d_in[5];
  float* Y = (float*)d_out;

  // workspace carve-up (all bf16): total ~48 MiB
  char* ws = (char*)d_ws;
  __bf16* Xb  = (__bf16*)ws; ws += (size_t)B_ * T_ * C_ * 2;       // 8 MiB
  __bf16* Wb  = (__bf16*)ws; ws += (size_t)3 * C_ * C_ * 2;        // 6 MiB
  __bf16* Wpb = (__bf16*)ws; ws += (size_t)C_ * C_ * 2;            // 2 MiB
  __bf16* Qb  = (__bf16*)ws; ws += (size_t)B_ * H_ * T_ * D_ * 2;  // 8 MiB
  __bf16* Kb  = (__bf16*)ws; ws += (size_t)B_ * H_ * T_ * D_ * 2;  // 8 MiB
  __bf16* VTb = (__bf16*)ws; ws += (size_t)B_ * H_ * T_ * D_ * 2;  // 8 MiB
  __bf16* Ob  = (__bf16*)ws; ws += (size_t)B_ * T_ * C_ * 2;       // 8 MiB

  mha_cvt_bf16<<<2048, 256, 0, stream>>>(X,  Xb,  B_ * T_ * C_);
  mha_cvt_bf16<<<1024, 256, 0, stream>>>(Wq, Wb,              C_ * C_);
  mha_cvt_bf16<<<1024, 256, 0, stream>>>(Wk, Wb +     C_ * C_, C_ * C_);
  mha_cvt_bf16<<<1024, 256, 0, stream>>>(Wv, Wb + 2 * C_ * C_, C_ * C_);
  mha_cvt_bf16<<<1024, 256, 0, stream>>>(Wp, Wpb,             C_ * C_);

  // QKV: 128 row tiles x 96 col tiles (8 waves/block, 32x32 per wave)
  mha_qkv_gemm<<<dim3(B_ * T_ / 32, 12), 256, 0, stream>>>(Xb, Wb, Qb, Kb, VTb);

  // flash attention: B*H*T/32 = 2048 wave-jobs, 8 per block
  mha_flash<<<dim3(256), 256, 0, stream>>>(Qb, Kb, VTb, Ob);

  // output projection: 128 row tiles x 32 col tiles (8 waves/block)
  mha_out_proj<<<dim3(B_ * T_ / 32, 4), 256, 0, stream>>>(Ob, Wpb, bp, Y);
}
